// TSPModel_90005334655065
// MI455X (gfx1250) — matrix operand order repaired and checked by reading, not compiled, not run on hardware
//
#include <hip/hip_runtime.h>

// ---------------------------------------------------------------------------
// Problem constants (from the reference): B=2, N=4096, E=128, H=8, D=16, F=512
// ---------------------------------------------------------------------------
#define NTOK 4096
#define EDIM 128
#define FDIM 512
#define ROWS 8192        // B*N
#define LOG2N 12.0f

typedef __attribute__((ext_vector_type(16))) __bf16 v16bf;
typedef __attribute__((ext_vector_type(8)))  float  v8f;

struct alignas(16) Chunk16 { unsigned int w[4]; };   // 16 bytes == 8 bf16
struct alignas(8)  BF4     { __bf16 h[4]; };

// Async global->LDS (CDNA5): guarded so a missing builtin falls back cleanly.
#if defined(__AMDGCN__) && __has_builtin(__builtin_amdgcn_global_load_async_to_lds_b128) && __has_builtin(__builtin_amdgcn_s_wait_asynccnt)
#define HAVE_ASYNC_LDS 1
typedef __attribute__((ext_vector_type(4))) int v4i;
typedef __attribute__((address_space(1))) v4i gas_v4i;   // global (AS1)
typedef __attribute__((address_space(3))) v4i las_v4i;   // LDS    (AS3)
#else
#define HAVE_ASYNC_LDS 0
#endif

static __device__ __forceinline__ v16bf ld_v16bf(const __bf16* p) {
  v16bf r;
  ((Chunk16*)&r)[0] = ((const Chunk16*)p)[0];
  ((Chunk16*)&r)[1] = ((const Chunk16*)p)[1];
  return r;
}

static __device__ __forceinline__ v8f wmma_bf16(v16bf a, v16bf b, v8f c) {
  // D = A(16x32 bf16) * B(32x16 bf16) + C(16x16 f32)
  return __builtin_amdgcn_wmma_f32_16x16x32_bf16(
      /*neg_a=*/false, a, /*neg_b=*/false, b,
      /*c_mod=*/(short)0, c, /*reuse_a=*/false, /*reuse_b=*/false);
}

// One 16x16 output tile of  C = A(row-major, lda) * Bt(col-major-as-rows, ldb)^T
static __device__ __forceinline__ v8f gemm_tile(
    const __bf16* __restrict__ A, int lda,
    const __bf16* __restrict__ Bt, int ldb,
    int K, int rowbase, int colbase, int lane)
{
  const int ln = lane & 15, half = lane >> 4;
  const __bf16* arow = A + (size_t)(rowbase + ln) * lda;
  const __bf16* brow = Bt + (size_t)(colbase + ln) * ldb + half * 8;
  v8f c = {};
  for (int k0 = 0; k0 < K; k0 += 32) {
    v16bf a;  // A 16x32 bf16: e0..7 -> K=half*8+e ; e8..15 -> K=16+half*8+(e-8)
    const __bf16* p0 = arow + k0 + half * 8;
    ((Chunk16*)&a)[0] = *(const Chunk16*)(p0);
    ((Chunk16*)&a)[1] = *(const Chunk16*)(p0 + 16);
    v16bf b;  // B 32x16: e0..7 -> K=half*8+e ; e8..15 -> K=16+half*8+(e-8)
    ((Chunk16*)&b)[0] = *(const Chunk16*)(brow + k0);
    ((Chunk16*)&b)[1] = *(const Chunk16*)(brow + k0 + 16);
    c = wmma_bf16(a, b, c);
  }
  return c;
}

// ---------------------------------------------------------------------------
// K0: fp32 -> bf16 weight conversion
// ---------------------------------------------------------------------------
__global__ void cvt_bf16_kernel(const float* __restrict__ s,
                                __bf16* __restrict__ d, int n) {
  int i = blockIdx.x * 256 + threadIdx.x;
  if (i < n) d[i] = (__bf16)s[i];
}

// ---------------------------------------------------------------------------
// K1/K6: RMS-norm over E=128, one wave (32 lanes x 4 elems) per row
// ---------------------------------------------------------------------------
__global__ void __launch_bounds__(256) rmsnorm_kernel(
    const float* __restrict__ x, const float* __restrict__ w,
    __bf16* __restrict__ o) {
  const int wv = threadIdx.x >> 5, lane = threadIdx.x & 31;
  const size_t row = (size_t)blockIdx.x * 8 + wv;
  const float4 xv = ((const float4*)(x + row * EDIM))[lane];
  float ss = xv.x * xv.x + xv.y * xv.y + xv.z * xv.z + xv.w * xv.w;
#pragma unroll
  for (int off = 16; off > 0; off >>= 1) ss += __shfl_xor(ss, off, 32);
  const float sc = rsqrtf(ss * (1.0f / 128.0f) + 1e-6f);
  const float4 wv4 = ((const float4*)w)[lane];
  BF4 r;
  r.h[0] = (__bf16)(xv.x * sc * wv4.x);
  r.h[1] = (__bf16)(xv.y * sc * wv4.y);
  r.h[2] = (__bf16)(xv.z * sc * wv4.z);
  r.h[3] = (__bf16)(xv.w * sc * wv4.w);
  *(BF4*)(o + row * EDIM + lane * 4) = r;
}

// ---------------------------------------------------------------------------
// K2: QKV projection (WMMA) + sigmoid(q) + [kw*v | kw] written TRANSPOSED
//     kvkwT layout: [B, 256, N] bf16  (cols 0..127 = kw*v, 128..255 = kw)
// ---------------------------------------------------------------------------
__global__ void __launch_bounds__(256) qkv_kernel(
    const __bf16* __restrict__ h, const __bf16* __restrict__ Wqb,
    const __bf16* __restrict__ Wkb, const __bf16* __restrict__ Wvb,
    float* __restrict__ qsig, __bf16* __restrict__ kvkwT) {
  const int lane = threadIdx.x & 31, wv = threadIdx.x >> 5;
  const int rb = blockIdx.x * 16;
  const int colb = wv * 16;
  const int ln = lane & 15, half = lane >> 4;
  v8f cq = gemm_tile(h, EDIM, Wqb, EDIM, EDIM, rb, colb, lane);
  v8f ck = gemm_tile(h, EDIM, Wkb, EDIM, EDIM, rb, colb, lane);
  v8f cv = gemm_tile(h, EDIM, Wvb, EDIM, EDIM, rb, colb, lane);
  const int col = colb + ln;
#pragma unroll
  for (int r = 0; r < 8; r++) {
    const int row = rb + r + half * 8;
    qsig[(size_t)row * EDIM + col] = 1.0f / (1.0f + __expf(-cq[r]));
    const float kw = __expf(ck[r]);
    const float kv = kw * cv[r];
    const int b = row >> 12, n = row & (NTOK - 1);
    kvkwT[((size_t)b * 256 + col) * NTOK + n]       = (__bf16)kv;
    kvkwT[((size_t)b * 256 + 128 + col) * NTOK + n] = (__bf16)kw;
  }
}

// ---------------------------------------------------------------------------
// K3: W12[b, j, 0..255] += sum_i exp(-c*dis[b,j,i]) * kvkwT[b,:,i]
// Block = 128 rows x 256 cols x 1024-wide K slice; 8 waves, each wave owns
// 16 rows x 256 cols (16 f32 accumulators). B slab double-buffered in LDS
// (async global->LDS when available). K-slices combined via f32 atomics.
// ---------------------------------------------------------------------------
__global__ void __launch_bounds__(256) attn_big_kernel(
    const float* __restrict__ dis, const __bf16* __restrict__ kvkwT,
    float* __restrict__ W12, const float* __restrict__ alpha_p) {
  const int b  = blockIdx.y;
  const int jb = blockIdx.x & 31;   // 32 row-blocks of 128
  const int is = blockIdx.x >> 5;   // 4 K-slices of 1024
  const int wv = threadIdx.x >> 5, lane = threadIdx.x & 31;
  const int ln = lane & 15, half = lane >> 4;
  const int jbase = jb * 128 + wv * 16;
  const float coef = alpha_p[0] * LOG2N;

  // double-buffered B slab; 80B row stride kills bank conflicts, keeps 16B align
  __shared__ alignas(16) __bf16 Bs[2][256][40];

  v8f acc[16];
#pragma unroll
  for (int t = 0; t < 16; t++) acc[t] = (v8f){};

  const float* disrow = dis + ((size_t)b * NTOK + jbase + ln) * NTOK;
  const int i_beg = is * 1024, i_end = i_beg + 1024;
  const int c = threadIdx.x;  // 256 threads <-> 256 cols
  const __bf16* colsrc = kvkwT + ((size_t)b * 256 + c) * NTOK;

  // --- slab fill: Bs[buf][c][0..31] = kvkwT[b, c, i0..i0+31] (64B / thread)
  auto fill = [&](int buf, int i0) {
#if HAVE_ASYNC_LDS
#pragma unroll
    for (int q = 0; q < 4; q++) {
      __builtin_amdgcn_global_load_async_to_lds_b128(
          (gas_v4i*)(colsrc + i0 + q * 8),
          (las_v4i*)(&Bs[buf][c][q * 8]),
          0, 0);
    }
#else
    const Chunk16* s = (const Chunk16*)(colsrc + i0);
    Chunk16* d = (Chunk16*)&Bs[buf][c][0];
    d[0] = s[0]; d[1] = s[1]; d[2] = s[2]; d[3] = s[3];
#endif
  };

  fill(0, i_beg);
#if HAVE_ASYNC_LDS
  __builtin_amdgcn_s_wait_asynccnt(0);
#endif
  __syncthreads();

  int cur = 0;
  for (int i0 = i_beg; i0 < i_end; i0 += 32) {
    // prefetch next slab into the other buffer while we compute on this one
    if (i0 + 32 < i_end) fill(cur ^ 1, i0 + 32);

    // ---- build A fragment: exp(-coef * dis) -> bf16, WMMA A 16x32 layout
    const float* p0 = disrow + i0 + half * 8;
    __builtin_prefetch(p0 + 32, 0, 0);   // stream the dis matrix
    float4 x0 = *(const float4*)(p0);
    float4 x1 = *(const float4*)(p0 + 4);
    float4 x2 = *(const float4*)(p0 + 16);
    float4 x3 = *(const float4*)(p0 + 20);
    float xs[16] = {x0.x, x0.y, x0.z, x0.w, x1.x, x1.y, x1.z, x1.w,
                    x2.x, x2.y, x2.z, x2.w, x3.x, x3.y, x3.z, x3.w};
    v16bf a;
#pragma unroll
    for (int e = 0; e < 16; e++) a[e] = (__bf16)__expf(-coef * xs[e]);

    // ---- 16 column tiles of WMMA against the current LDS slab
#pragma unroll
    for (int t = 0; t < 16; t++) {
      v16bf bb;
      ((Chunk16*)&bb)[0] = *(const Chunk16*)(&Bs[cur][t * 16 + ln][half * 8]);
      ((Chunk16*)&bb)[1] = *(const Chunk16*)(&Bs[cur][t * 16 + ln][half * 8 + 16]);
      acc[t] = wmma_bf16(a, bb, acc[t]);
    }

#if HAVE_ASYNC_LDS
    __builtin_amdgcn_s_wait_asynccnt(0);  // own async fill of nxt complete
#endif
    __syncthreads();                      // all waves done filling + reading
    cur ^= 1;
  }

  // ---- epilogue: atomic accumulate the K-slice partials
  float* outbase = W12 + ((size_t)b * NTOK + jbase) * 256;
#pragma unroll
  for (int t = 0; t < 16; t++) {
    const int col = t * 16 + ln;
#pragma unroll
    for (int r = 0; r < 8; r++) {
      atomicAdd(outbase + (size_t)(r + half * 8) * 256 + col, acc[t][r]);
    }
  }
}

// ---------------------------------------------------------------------------
// K4: attn = sigmoid(q) * w1 / w2   -> bf16 row-major [ROWS, 128]
// ---------------------------------------------------------------------------
__global__ void attn_ew_kernel(const float* __restrict__ qsig,
                               const float* __restrict__ W12,
                               __bf16* __restrict__ attn) {
  const int idx = blockIdx.x * 256 + threadIdx.x;  // row*128 + col
  const int row = idx >> 7, col = idx & 127;
  const size_t base = (size_t)row * 256;
  const float w1 = W12[base + col];
  const float w2 = W12[base + 128 + col];
  attn[idx] = (__bf16)(qsig[idx] * (w1 / w2));
}

// ---------------------------------------------------------------------------
// K5: out1 = input1 + attn @ Wcomb.T
// ---------------------------------------------------------------------------
__global__ void __launch_bounds__(256) comb_kernel(
    const __bf16* __restrict__ attn, const __bf16* __restrict__ Wcb,
    const float* __restrict__ input1, float* __restrict__ out1) {
  const int lane = threadIdx.x & 31, wv = threadIdx.x >> 5;
  const int rb = blockIdx.x * 16, colb = wv * 16;
  const int ln = lane & 15, half = lane >> 4;
  v8f c = gemm_tile(attn, EDIM, Wcb, EDIM, EDIM, rb, colb, lane);
  const int col = colb + ln;
#pragma unroll
  for (int r = 0; r < 8; r++) {
    const size_t idx = (size_t)(rb + r + half * 8) * EDIM + col;
    out1[idx] = input1[idx] + c[r];
  }
}

// ---------------------------------------------------------------------------
// K7: mid = relu(h2 @ W1.T + b1)  -> bf16 [ROWS, 512]
// ---------------------------------------------------------------------------
__global__ void __launch_bounds__(256) ffn1_kernel(
    const __bf16* __restrict__ h2, const __bf16* __restrict__ W1b,
    const float* __restrict__ b1, __bf16* __restrict__ mid) {
  const int lane = threadIdx.x & 31, wv = threadIdx.x >> 5;
  const int rb = blockIdx.x * 16;
  const int colb = blockIdx.y * 128 + wv * 16;
  const int ln = lane & 15, half = lane >> 4;
  v8f c = gemm_tile(h2, EDIM, W1b, EDIM, EDIM, rb, colb, lane);
  const int col = colb + ln;
  const float bias = b1[col];
#pragma unroll
  for (int r = 0; r < 8; r++) {
    const int row = rb + r + half * 8;
    mid[(size_t)row * FDIM + col] = (__bf16)fmaxf(c[r] + bias, 0.0f);
  }
}

// ---------------------------------------------------------------------------
// K8: out = out1 + mid @ W2.T + b2
// ---------------------------------------------------------------------------
__global__ void __launch_bounds__(256) ffn2_kernel(
    const __bf16* __restrict__ mid, const __bf16* __restrict__ W2b,
    const float* __restrict__ b2, const float* __restrict__ out1,
    float* __restrict__ out) {
  const int lane = threadIdx.x & 31, wv = threadIdx.x >> 5;
  const int rb = blockIdx.x * 16, colb = wv * 16;
  const int ln = lane & 15, half = lane >> 4;
  v8f c = gemm_tile(mid, FDIM, W2b, FDIM, FDIM, rb, colb, lane);
  const int col = colb + ln;
  const float bias = b2[col];
#pragma unroll
  for (int r = 0; r < 8; r++) {
    const size_t idx = (size_t)(rb + r + half * 8) * EDIM + col;
    out[idx] = out1[idx] + c[r] + bias;
  }
}

// ---------------------------------------------------------------------------
// Workspace layout (bytes)
// ---------------------------------------------------------------------------
static const size_t OFF_WQ  = 0;
static const size_t OFF_WK  = OFF_WQ  + (size_t)16384 * 2;
static const size_t OFF_WV  = OFF_WK  + (size_t)16384 * 2;
static const size_t OFF_WC  = OFF_WV  + (size_t)16384 * 2;
static const size_t OFF_W1  = OFF_WC  + (size_t)16384 * 2;
static const size_t OFF_W2  = OFF_W1  + (size_t)65536 * 2;
static const size_t OFF_H   = OFF_W2  + (size_t)65536 * 2;          // bf16 [8192,128]
static const size_t OFF_QS  = OFF_H   + (size_t)ROWS * EDIM * 2;    // f32  [8192,128]
static const size_t OFF_KV  = OFF_QS  + (size_t)ROWS * EDIM * 4;    // bf16 [2,256,4096]
static const size_t OFF_W12 = OFF_KV  + (size_t)2 * 256 * NTOK * 2; // f32  [2,4096,256]
static const size_t OFF_AT  = OFF_W12 + (size_t)2 * NTOK * 256 * 4; // bf16 [8192,128]
static const size_t OFF_O1  = OFF_AT  + (size_t)ROWS * EDIM * 2;    // f32  [8192,128]
static const size_t OFF_H2  = OFF_O1  + (size_t)ROWS * EDIM * 4;    // bf16 [8192,128]
static const size_t OFF_MID = OFF_H2  + (size_t)ROWS * EDIM * 2;    // bf16 [8192,512]

extern "C" void kernel_launch(void* const* d_in, const int* in_sizes, int n_in,
                              void* d_out, int out_size, void* d_ws,
                              size_t ws_size, hipStream_t stream) {
  (void)in_sizes; (void)n_in; (void)out_size; (void)ws_size;
  const float* input1 = (const float*)d_in[0];
  const float* dis    = (const float*)d_in[1];
  const float* Wq     = (const float*)d_in[2];
  const float* Wk     = (const float*)d_in[3];
  const float* Wv     = (const float*)d_in[4];
  const float* alpha  = (const float*)d_in[5];
  const float* ln1w   = (const float*)d_in[6];
  const float* ln2w   = (const float*)d_in[7];
  const float* Wcomb  = (const float*)d_in[8];
  const float* W1     = (const float*)d_in[9];
  const float* b1     = (const float*)d_in[10];
  const float* W2     = (const float*)d_in[11];
  const float* b2     = (const float*)d_in[12];
  float* out = (float*)d_out;

  char* ws = (char*)d_ws;
  __bf16* Wq_bf  = (__bf16*)(ws + OFF_WQ);
  __bf16* Wk_bf  = (__bf16*)(ws + OFF_WK);
  __bf16* Wv_bf  = (__bf16*)(ws + OFF_WV);
  __bf16* Wc_bf  = (__bf16*)(ws + OFF_WC);
  __bf16* W1_bf  = (__bf16*)(ws + OFF_W1);
  __bf16* W2_bf  = (__bf16*)(ws + OFF_W2);
  __bf16* h_bf   = (__bf16*)(ws + OFF_H);
  float*  qsig   = (float*) (ws + OFF_QS);
  __bf16* kvkwT  = (__bf16*)(ws + OFF_KV);
  float*  W12    = (float*) (ws + OFF_W12);
  __bf16* attn   = (__bf16*)(ws + OFF_AT);
  float*  out1   = (float*) (ws + OFF_O1);
  __bf16* h2_bf  = (__bf16*)(ws + OFF_H2);
  __bf16* mid_bf = (__bf16*)(ws + OFF_MID);

  // K0: weights -> bf16 (layouts already match the WMMA B^T operand)
  cvt_bf16_kernel<<<64,  256, 0, stream>>>(Wq,    Wq_bf, 16384);
  cvt_bf16_kernel<<<64,  256, 0, stream>>>(Wk,    Wk_bf, 16384);
  cvt_bf16_kernel<<<64,  256, 0, stream>>>(Wv,    Wv_bf, 16384);
  cvt_bf16_kernel<<<64,  256, 0, stream>>>(Wcomb, Wc_bf, 16384);
  cvt_bf16_kernel<<<256, 256, 0, stream>>>(W1,    W1_bf, 65536);
  cvt_bf16_kernel<<<256, 256, 0, stream>>>(W2,    W2_bf, 65536);

  // K1: h = rmsnorm(input1) * ln1_w
  rmsnorm_kernel<<<ROWS / 8, 256, 0, stream>>>(input1, ln1w, h_bf);

  // K2: QKV (WMMA) + sigmoid / exp, transposed kv|kw operand
  qkv_kernel<<<ROWS / 16, 256, 0, stream>>>(h_bf, Wq_bf, Wk_bf, Wv_bf,
                                            qsig, kvkwT);

  // K3: dis-weighted aggregation (the 17-GFLOP matmul)
  (void)hipMemsetAsync(W12, 0, (size_t)2 * NTOK * 256 * sizeof(float), stream);
  attn_big_kernel<<<dim3(128, 2), 256, 0, stream>>>(dis, kvkwT, W12, alpha);

  // K4: attn = sigmoid(q) * w1/w2
  attn_ew_kernel<<<(ROWS * EDIM) / 256, 256, 0, stream>>>(qsig, W12, attn);

  // K5: out1 = input1 + attn @ Wcomb.T
  comb_kernel<<<ROWS / 16, 256, 0, stream>>>(attn, Wc_bf, input1, out1);

  // K6: h2 = rmsnorm(out1) * ln2_w
  rmsnorm_kernel<<<ROWS / 8, 256, 0, stream>>>(out1, ln2w, h2_bf);

  // K7/K8: FFN
  ffn1_kernel<<<dim3(ROWS / 16, 4), 256, 0, stream>>>(h2_bf, W1_bf, b1, mid_bf);
  ffn2_kernel<<<ROWS / 16, 256, 0, stream>>>(mid_bf, W2_bf, b2, out1, out);
}
